// Block_60601988546807
// MI455X (gfx1250) — matrix operand order repaired and checked by reading, not compiled, not run on hardware
//
#include <hip/hip_runtime.h>

// ---------------------------------------------------------------------------
// StyleGAN2-style modulated conv block on MI455X (gfx1250):
//   softmax kernel mixing -> modulate -> demodulate -> per-sample 3x3 conv
//   (implicit GEMM, v_wmma_f32_16x16x32_f16, TDM-staged A, double-buffered B)
//   -> GroupNorm(8) -> SiLU
// ---------------------------------------------------------------------------

typedef __attribute__((ext_vector_type(16))) _Float16 v16h;
typedef __attribute__((ext_vector_type(8)))  float    v8f;
typedef __attribute__((ext_vector_type(4)))  unsigned int v4u;
typedef __attribute__((ext_vector_type(8)))  int      v8i;
typedef __attribute__((ext_vector_type(4)))  int      v4i;

#define HH   256
#define WW   256
#define HW_  65536         // 256*256
#define KTOT 576           // 9 taps * 64 channels
#define NKC  18            // 576 / 32 K-chunks
#define GN_N 524288.0f     // 8 ch * 65536 px per (b, group)

#define A_HALVES_PER_B (4 * NKC * 512)            // 36,864 halves = 73,728 B
#define AW_BYTES   (16 * A_HALVES_PER_B * 2)      // 1,179,648 B of f16 A-fragments
#define STATS_OFF  AW_BYTES                       // 256 floats (sum, sumsq)
#define MR_OFF     (AW_BYTES + 1024)              // 256 floats (mean, rstd)

#define BSLOT 24                                  // 48B-padded fragment slot (halves)
#define BBUF  (256 * BSLOT)                       // one B buffer (halves)

// ---------------------------------------------------------------------------
// Kernel 0: zero the GN partial-stat accumulators (ws is poisoned by harness)
// ---------------------------------------------------------------------------
__global__ void zero_stats(float* stats) {
    stats[threadIdx.x] = 0.0f;     // 256 threads, 256 floats
}

// ---------------------------------------------------------------------------
// Kernel 1: per-batch weight prep (f32 mixing/modulation/demod, f16 output
// pre-swizzled into the exact WMMA 16x32 A-fragment layout, K = (tap, ch)).
// ---------------------------------------------------------------------------
__global__ __launch_bounds__(256) void wprep(const float* __restrict__ mod,
                                             const float* __restrict__ kmod,
                                             const float* __restrict__ cw,
                                             _Float16* __restrict__ Aw) {
    const int o = blockIdx.x;
    const int b = blockIdx.y;
    const int t = threadIdx.x;

    float k0 = kmod[b * 4 + 0], k1 = kmod[b * 4 + 1];
    float k2 = kmod[b * 4 + 2], k3 = kmod[b * 4 + 3];
    float m = fmaxf(fmaxf(k0, k1), fmaxf(k2, k3));
    float a0 = __expf(k0 - m), a1 = __expf(k1 - m);
    float a2 = __expf(k2 - m), a3 = __expf(k3 - m);
    float inv_s = 1.0f / (a0 + a1 + a2 + a3);
    a0 *= inv_s; a1 *= inv_s; a2 *= inv_s; a3 *= inv_s;

    float wv[3];
    float ssq = 0.0f;
#pragma unroll
    for (int r = 0; r < 3; ++r) {
        int id = t + r * 256;          // K index = tap*64 + i
        float w = 0.0f;
        if (id < KTOT) {
            int tap = id >> 6, i = id & 63;
            int kk = tap / 3, ll = tap % 3;
            int ci = ((o * 64 + i) * 9) + kk * 3 + ll;   // [n,o,i,3,3], n-stride 36864
            float acc = a0 * cw[ci] + a1 * cw[ci + 36864]
                      + a2 * cw[ci + 73728] + a3 * cw[ci + 110592];
            w = acc * (mod[b * 64 + i] + 1.0f);
        }
        wv[r] = w;
        ssq += w * w;
    }

#pragma unroll
    for (int off = 16; off >= 1; off >>= 1) ssq += __shfl_xor(ssq, off, 32);
    __shared__ float red[8];
    if ((t & 31) == 0) red[t >> 5] = ssq;
    __syncthreads();
    float tot = red[0] + red[1] + red[2] + red[3] + red[4] + red[5] + red[6] + red[7];
    float inv = rsqrtf(fmaxf(tot, 1e-8f));

#pragma unroll
    for (int r = 0; r < 3; ++r) {
        int id = t + r * 256;
        if (id < KTOT) {
            _Float16 h = (_Float16)(wv[r] * inv);
            int kc = id >> 5;
            int rr = id & 31;
            // inverse of A-frag layout: K(lane,e) = e + 8*(e>=8) + 8*(lane>=16)
            int lane, e;
            if (rr < 8)       { lane = 0;  e = rr; }
            else if (rr < 16) { lane = 16; e = rr - 8; }
            else if (rr < 24) { lane = 0;  e = rr - 8; }
            else              { lane = 16; e = rr - 16; }
            lane += (o & 15);
            Aw[(((b * 4 + (o >> 4)) * NKC + kc) * 512) + lane * 16 + e] = h;
        }
    }
}

// ---------------------------------------------------------------------------
// Kernel 2: implicit-GEMM conv. Workgroup = 64 out-ch x 128 px, 8 waves.
//   * A (73,728 B, fragment-linear) DMA'd once per WG via tensor_load_to_lds
//   * B staged per K-chunk through a double-buffered 48B-padded LDS layout
//   * software-pipelined: 1 barrier / chunk, next chunk's loads overlap WMMA
//   * fused GroupNorm partial sums via f32 atomics
// ---------------------------------------------------------------------------
__global__ __launch_bounds__(256) void conv_wmma(const float* __restrict__ x,
                                                 const _Float16* __restrict__ Aw,
                                                 float* __restrict__ y,
                                                 float* __restrict__ stats) {
    __shared__ __align__(32) _Float16 ldsA[A_HALVES_PER_B];   // 73,728 B
    __shared__ __align__(32) _Float16 ldsB[2 * BBUF];         // 24,576 B

    const int t   = threadIdx.x;
    const int b   = blockIdx.y;
    const int tl  = blockIdx.x;           // 512 tiles per batch
    const int row = tl >> 1;              // output row
    const int px0 = (tl & 1) << 7;        // 0 or 128

    const int l0 = t & 31;                // lane
    const int wv = t >> 5;                // wave id 0..7
    const int mt = wv >> 1;               // M-tile (16 channels)
    const int nh = wv & 1;                // pixel half (64 px)

    // staging role: thread t fills fragment slot t (subtile t>>5, lane t&31)
    const int cc     = ((t >> 5) << 4) | (l0 & 15);   // column 0..127
    const int kkbase = (l0 & 16) >> 1;                // K offset 0 or 8

    auto stage_load = [&](int kc) -> v16h {
        const int tap  = kc >> 1;
        const int ch0  = (kc & 1) << 5;
        const int irow = row + (tap / 3) - 1;
        const int icol = px0 + cc + (tap % 3) - 1;
        const bool valid = ((unsigned)irow < 256u) && ((unsigned)icol < 256u);
        const float* xp = x + (((size_t)(b * 64 + ch0 + kkbase) * HH + (irow & 255)) * WW
                               + (icol & 255));
        v16h bv = {};
        if (valid) {
#pragma unroll
            for (int e = 0; e < 16; ++e) bv[e] = (_Float16)xp[(e + (e & 8)) * HW_];
        }
        return bv;
    };

    // ---- prologue: DMA the whole A block for this batch into LDS (wave 0),
    //      and stage B chunk 0 into buffer 0 -------------------------------
    {
        v16h c0 = stage_load(0);
        *(v16h*)&ldsB[t * BSLOT] = c0;
    }
    if (wv == 0) {
        // Tensor DMA descriptor (ISA 8.3/8.4): 1-D copy of 18432 dwords.
        unsigned long long ga = (unsigned long long)(const void*)(Aw + (size_t)b * A_HALVES_PER_B);
        unsigned la = (unsigned)(size_t)(void*)&ldsA[0];      // generic->LDS offset (addr[31:0])
        v4u g0;
        g0[0] = 1u;                                           // count=1, no gather
        g0[1] = la;                                           // lds_addr
        g0[2] = (unsigned)(ga & 0xFFFFFFFFu);                 // global_addr[31:0]
        g0[3] = (unsigned)((ga >> 32) & 0x01FFFFFFu) | 0x80000000u;  // addr[56:32] | type=2
        v8i g1;
        g1[0] = 0x00020000;                                   // data_size=4B, mask=0
        g1[1] = (int)(0x4800u << 16);                         // tensor_dim0 = 18432 (lo16)
        g1[2] = 0x00010000;                                   // tensor_dim1 = 1
        g1[3] = (int)(0x4800u << 16);                         // tile_dim0 = 18432
        g1[4] = 0;                                            // tile_dim1/2 = 0
        g1[5] = 0x4800;                                       // tensor_dim0_stride = 18432
        g1[6] = 0; g1[7] = 0;
        v4i g2 = {0, 0, 0, 0};
        v4i g3 = {0, 0, 0, 0};
        v8i g4 = {0, 0, 0, 0, 0, 0, 0, 0};                    // unused trailing group (clang-23 6-arg form)
        __builtin_amdgcn_tensor_load_to_lds(g0, g1, g2, g3, g4, 0);
        __builtin_amdgcn_s_wait_tensorcnt(0);                 // TENSORcnt==0 for wave 0
    }
    __syncthreads();                                          // publish A + B0 to all waves

    v8f acc[4] = {};

    for (int kc = 0; kc < NKC; ++kc) {
        // issue next chunk's global loads early (overlap with WMMA below)
        v16h nxt = {};
        if (kc + 1 < NKC) nxt = stage_load(kc + 1);

        // A fragment from LDS (placed by TDM)
        v16h av = *(const v16h*)&ldsA[((mt * NKC + kc) * 32 + l0) * 16];

        const int rb = (kc & 1) * BBUF;
#pragma unroll
        for (int si = 0; si < 4; ++si) {
            v16h bf = *(const v16h*)&ldsB[rb + ((nh * 4 + si) * 32 + l0) * BSLOT];
            acc[si] = __builtin_amdgcn_wmma_f32_16x16x32_f16(
                false, av, false, bf, (short)0, acc[si], false, false);
        }

        if (kc + 1 < NKC)
            *(v16h*)&ldsB[((kc + 1) & 1) * BBUF + t * BSLOT] = nxt;
        __syncthreads();
    }

    // D layout: vgpr r, lane n(<16): row r col n; lane n+16: row r+8 col n
    float lsum = 0.0f, lsq = 0.0f;
    const int chbase  = mt * 16 + ((l0 >= 16) ? 8 : 0);
    const int colbase = px0 + nh * 64 + (l0 & 15);
#pragma unroll
    for (int si = 0; si < 4; ++si) {
        const int col = colbase + si * 16;
#pragma unroll
        for (int r = 0; r < 8; ++r) {
            float v = acc[si][r];
            y[((size_t)(b * 64 + chbase + r) * HH + row) * WW + col] = v;
            lsum += v;
            lsq  += v * v;
        }
    }

    // lanes 0-15 -> group 2*mt, lanes 16-31 -> group 2*mt+1
#pragma unroll
    for (int off = 8; off >= 1; off >>= 1) {
        lsum += __shfl_xor(lsum, off, 16);
        lsq  += __shfl_xor(lsq,  off, 16);
    }
    if ((l0 & 15) == 0) {
        const int g = mt * 2 + (l0 >> 4);
        atomicAdd(&stats[(b * 8 + g) * 2 + 0], lsum);
        atomicAdd(&stats[(b * 8 + g) * 2 + 1], lsq);
    }
}

// ---------------------------------------------------------------------------
// Kernel 3: finalize GN stats: mean, rstd per (b, group)
// ---------------------------------------------------------------------------
__global__ void gn_finalize(const float* __restrict__ stats, float* __restrict__ mr) {
    const int i = threadIdx.x;    // 0..127
    if (i < 128) {
        const float invN = 1.0f / GN_N;
        float mean = stats[i * 2] * invN;
        float var  = stats[i * 2 + 1] * invN - mean * mean;
        mr[i * 2]     = mean;
        mr[i * 2 + 1] = rsqrtf(var + 1e-5f);
    }
}

// ---------------------------------------------------------------------------
// Kernel 4: GN apply + SiLU, in place over d_out, float4-vectorized.
// ---------------------------------------------------------------------------
__global__ __launch_bounds__(256) void gn_silu(float* __restrict__ y,
                                               const float* __restrict__ mr,
                                               const float* __restrict__ gamma,
                                               const float* __restrict__ beta) {
    const unsigned idx = blockIdx.x * 256u + threadIdx.x;    // float4 index
    const unsigned plane = idx >> 14;                        // 16384 float4 per plane
    const int ch = plane & 63;
    const int b  = plane >> 6;
    const int g  = ch >> 3;
    const float mean = mr[(b * 8 + g) * 2];
    const float rstd = mr[(b * 8 + g) * 2 + 1];
    const float sc = gamma[ch] * rstd;
    const float sh = beta[ch] - mean * sc;

    float4 v = ((float4*)y)[idx];
    float u0 = v.x * sc + sh, u1 = v.y * sc + sh;
    float u2 = v.z * sc + sh, u3 = v.w * sc + sh;
    v.x = u0 / (1.0f + __expf(-u0));
    v.y = u1 / (1.0f + __expf(-u1));
    v.z = u2 / (1.0f + __expf(-u2));
    v.w = u3 / (1.0f + __expf(-u3));
    ((float4*)y)[idx] = v;
}

// ---------------------------------------------------------------------------
extern "C" void kernel_launch(void* const* d_in, const int* in_sizes, int n_in,
                              void* d_out, int out_size, void* d_ws, size_t ws_size,
                              hipStream_t stream) {
    (void)in_sizes; (void)n_in; (void)out_size; (void)ws_size;
    const float* x     = (const float*)d_in[0];
    const float* mod   = (const float*)d_in[1];
    const float* kmod  = (const float*)d_in[2];
    const float* cw    = (const float*)d_in[3];
    const float* gamma = (const float*)d_in[4];
    const float* beta  = (const float*)d_in[5];
    float* out = (float*)d_out;

    char* ws = (char*)d_ws;
    _Float16* Aw  = (_Float16*)ws;
    float* stats  = (float*)(ws + STATS_OFF);
    float* mr     = (float*)(ws + MR_OFF);

    zero_stats<<<1, 256, 0, stream>>>(stats);
    wprep<<<dim3(64, 16), 256, 0, stream>>>(mod, kmod, cw, Aw);
    conv_wmma<<<dim3(512, 16), 256, 0, stream>>>(x, Aw, out, stats);
    gn_finalize<<<1, 128, 0, stream>>>(stats, mr);
    gn_silu<<<65536, 256, 0, stream>>>(out, mr, gamma, beta);
}
